// WaveNetModel_36275293782603
// MI455X (gfx1250) — matrix operand order repaired and checked by reading, not compiled, not run on hardware
//
#include <hip/hip_runtime.h>
#include <hip/hip_bf16.h>
#include <math.h>

// MI455X / gfx1250: wave32, WMMA 16x16x32 f16 -> f32 accum.
// WaveNet as a chain of fused per-layer kernels in [n][t][c] f16 layout.
// One wave owns a 16-timestep column: computes the full 64-channel gated
// activation in registers, restripes it through a private 2KB LDS tile
// (D-layout -> B-layout), then runs the res (64x64) and skip (256x64) GEMMs
// from the same LDS tile. 1 kernel per layer instead of 3; gated tensor
// never touches global memory.

typedef __attribute__((ext_vector_type(16))) _Float16 v16h;
typedef __attribute__((ext_vector_type(8)))  _Float16 v8h;
typedef __attribute__((ext_vector_type(8)))  float    v8f;

#define TCAP 8192   // max time length (layer 0)
#define WN   2      // batch
#define WAVES_PER_BLK 8

// ---- WMMA fragment helpers (contiguous k: lane half h holds k = 16h + s) ----------

// A: 16x32 f16, row-major weights W[o][c], ld = row stride in halves.
static __device__ __forceinline__ v16h load_a_frag(const _Float16* __restrict__ W,
                                                   int ld, int m0, int kc, int lane) {
  const _Float16* p = W + (size_t)(m0 + (lane & 15)) * ld + kc + ((lane & 16) ? 16 : 0);
  return *(const v16h*)p;
}

// B: 32x16 f16 (K x N), activations Act[t][c]; row clamped so edge tiles
// never fault (stores are masked instead).
static __device__ __forceinline__ v16h load_b_frag(const _Float16* __restrict__ Act,
                                                   int ld, int t0, int kc, int lane, int tcap) {
  int t = t0 + (lane & 15);
  if (t > tcap - 1) t = tcap - 1;
  return *(const v16h*)(Act + (size_t)t * ld + kc + ((lane & 16) ? 16 : 0));
}

static __device__ __forceinline__ v8f wmma_f16(v16h a, v16h b, v8f c) {
  return __builtin_amdgcn_wmma_f32_16x16x32_f16(false, a, false, b, (short)0, c, false, false);
}

static __device__ __forceinline__ int d_obase(int m0, int lane) {
  return m0 + ((lane & 16) ? 8 : 0);          // D: 8 consecutive channels per lane
}

// ---- elementwise / conversion kernels -------------------------------------

__global__ void k_convert(const float* __restrict__ s, _Float16* __restrict__ d, int n) {
  int i = blockIdx.x * blockDim.x + threadIdx.x;
  if (i < n) d[i] = (_Float16)s[i];
}

// filter/gate: src [L][64][64][2] -> dst [L][2][64][64] (tap-major, [o][c] per tap)
__global__ void k_convert_tap(const float* __restrict__ s, _Float16* __restrict__ d, int n) {
  int i = blockIdx.x * blockDim.x + threadIdx.x;
  if (i >= n) return;
  int c = i & 63, o = (i >> 6) & 63, tap = (i >> 12) & 1, l = i >> 13;
  d[i] = (_Float16)s[(((l * 64 + o) * 64 + c) << 1) + tap];
}

// x (N,256,8192) f32 -> xT [n][t][256] f16
__global__ void k_xpose(const float* __restrict__ x, _Float16* __restrict__ xT) {
  int i = blockIdx.x * blockDim.x + threadIdx.x;
  if (i >= WN * 256 * TCAP) return;
  int t = i % TCAP, c = (i / TCAP) % 256, n = i / (TCAP * 256);
  xT[((size_t)n * TCAP + t) * 256 + c] = (_Float16)x[i];
}

__global__ void k_zero(float* __restrict__ p, int n) {
  int i = blockIdx.x * blockDim.x + threadIdx.x;
  if (i < n) p[i] = 0.0f;
}

__global__ void k_relu_h(const float* __restrict__ s, _Float16* __restrict__ d, int n) {
  int i = blockIdx.x * blockDim.x + threadIdx.x;
  if (i < n) { float v = s[i]; d[i] = (_Float16)(v > 0.0f ? v : 0.0f); }
}

// d_out[(n*240+oc)*1024 + ti] = tmp[(n*1024+ti)*240 + oc] + b2[oc]
__global__ void k_final(const float* __restrict__ tmp, const float* __restrict__ b2,
                        float* __restrict__ out) {
  int i = blockIdx.x * blockDim.x + threadIdx.x;
  if (i >= WN * 240 * 1024) return;
  int ti = i & 1023, oc = (i >> 10) % 240, n = i / (240 * 1024);
  out[i] = tmp[((size_t)n * 1024 + ti) * 240 + oc] + b2[oc];
}

// ---- start conv: h[t][o] = sum_c xT[t][c]*Ws[o][c]  (M=64 blocked, K=256) --

__global__ void __launch_bounds__(256) k_start(const _Float16* __restrict__ xT,
                                               const _Float16* __restrict__ Ws,
                                               _Float16* __restrict__ h) {
  int wid = (blockIdx.x * blockDim.x + threadIdx.x) >> 5;
  int lane = threadIdx.x & 31;
  const int TT = TCAP / 16;
  if (wid >= WN * TT) return;
  int n = wid / TT, tt = wid - n * TT;
  const _Float16* B = xT + (size_t)n * TCAP * 256;
  v8f acc[4];
#pragma unroll
  for (int j = 0; j < 4; ++j) acc[j] = (v8f){};
#pragma unroll
  for (int kc = 0; kc < 256; kc += 32) {
    v16h b = load_b_frag(B, 256, tt * 16, kc, lane, TCAP);
#pragma unroll
    for (int j = 0; j < 4; ++j)
      acc[j] = wmma_f16(load_a_frag(Ws, 256, j * 16, kc, lane), b, acc[j]);
  }
  int t = tt * 16 + (lane & 15);
#pragma unroll
  for (int j = 0; j < 4; ++j) {
    v8h o;
#pragma unroll
    for (int i = 0; i < 8; ++i) o[i] = (_Float16)acc[j][i];
    *(v8h*)(h + ((size_t)n * TCAP + t) * 64 + d_obase(j * 16, lane)) = o;
  }
}

// ---- fused layer kernel ---------------------------------------------------
// Per wave: 16-t column. Phase 1: gated = tanh(Wf*h)*sigmoid(Wg*h) (dilated
// K=2). Phase 2: restripe gated through LDS, hnext = Wr*gated + h[t+d].
// Phase 3 (last-1024 window only): skip += Wsk*gated.

__global__ void __launch_bounds__(256) k_layer(
    const _Float16* __restrict__ h, const _Float16* __restrict__ Wf,
    const _Float16* __restrict__ Wg, const _Float16* __restrict__ Wr,
    const _Float16* __restrict__ Wsk, _Float16* __restrict__ hnext,
    float* __restrict__ skip, int d, int Tout, int TT, int t_start) {
  __shared__ __align__(32) _Float16 lds[WAVES_PER_BLK][16 * 64];
  int wid = (blockIdx.x * blockDim.x + threadIdx.x) >> 5;
  int wv = threadIdx.x >> 5;
  int lane = threadIdx.x & 31;
  if (wid >= WN * TT) return;
  int n = wid / TT, tt = wid - n * TT;
  int t0 = tt * 16;
  int tl = lane & 15;
  const _Float16* B = h + (size_t)n * TCAP * 64;

  // -- phase 1: gating GEMMs, full M=64 in registers --
  v8f aF[4], aG[4];
#pragma unroll
  for (int j = 0; j < 4; ++j) { aF[j] = (v8f){}; aG[j] = (v8f){}; }
#pragma unroll
  for (int tap = 0; tap < 2; ++tap) {
    const _Float16* wf = Wf + tap * 64 * 64;
    const _Float16* wg = Wg + tap * 64 * 64;
#pragma unroll
    for (int kc = 0; kc < 64; kc += 32) {
      v16h b = load_b_frag(B, 64, t0 + tap * d, kc, lane, TCAP);
#pragma unroll
      for (int j = 0; j < 4; ++j) {
        aF[j] = wmma_f16(load_a_frag(wf, 64, j * 16, kc, lane), b, aF[j]);
        aG[j] = wmma_f16(load_a_frag(wg, 64, j * 16, kc, lane), b, aG[j]);
      }
    }
  }

  // activation; write 16x64 gated tile into this wave's private LDS region
  _Float16* tile = lds[wv];
#pragma unroll
  for (int j = 0; j < 4; ++j) {
    v8h o;
#pragma unroll
    for (int i = 0; i < 8; ++i) {
      float f = tanhf(aF[j][i]);
      float g = 1.0f / (1.0f + expf(-aG[j][i]));
      o[i] = (_Float16)(f * g);
    }
    *(v8h*)(tile + tl * 64 + d_obase(j * 16, lane)) = o;
  }
  // same-wave LDS ops are in-order; the wait + clobber keeps the compiler
  // from hoisting the B-layout reloads above the D-layout stores.
  asm volatile("s_wait_dscnt 0x0" ::: "memory");

  // reload gated as B fragments (K = channel dim), kept for phases 2 & 3
  v16h g0  = *(const v16h*)(tile + tl * 64 + ((lane & 16) ? 16 : 0));
  v16h g32 = *(const v16h*)(tile + tl * 64 + 32 + ((lane & 16) ? 16 : 0));

  int t = t0 + tl;
  bool tvalid = t < Tout;

  // -- phase 2: res conv + residual add --
#pragma unroll
  for (int j = 0; j < 4; ++j) {
    v8f acc = (v8f){};
    acc = wmma_f16(load_a_frag(Wr, 64, j * 16, 0, lane), g0, acc);
    acc = wmma_f16(load_a_frag(Wr, 64, j * 16, 32, lane), g32, acc);
    if (tvalid) {
      int ob = d_obase(j * 16, lane);
      v8h prev = *(const v8h*)(B + ((size_t)t + d) * 64 + ob);
      v8h o;
#pragma unroll
      for (int i = 0; i < 8; ++i) o[i] = (_Float16)(acc[i] + (float)prev[i]);
      *(v8h*)(hnext + ((size_t)n * TCAP + t) * 64 + ob) = o;
    }
  }

  // -- phase 3: skip conv, only tiles overlapping the last-1024 window --
  if (t0 + 15 >= t_start) {
    bool svalid = tvalid && (t >= t_start);
#pragma unroll
    for (int j = 0; j < 16; ++j) {
      v8f acc = (v8f){};
      acc = wmma_f16(load_a_frag(Wsk, 64, j * 16, 0, lane), g0, acc);
      acc = wmma_f16(load_a_frag(Wsk, 64, j * 16, 32, lane), g32, acc);
      if (svalid) {
        float* dst = skip + ((size_t)n * 1024 + (t - t_start)) * 256 + d_obase(j * 16, lane);
#pragma unroll
        for (int i = 0; i < 8; ++i) dst[i] += acc[i];
      }
    }
  }
}

// ---- end1: z2 = relu(z1 @ W^T + bias)  (M=256 in 4-tile groups, K=256) ----

__global__ void __launch_bounds__(256) k_end1(const _Float16* __restrict__ z1,
                                              const _Float16* __restrict__ W,
                                              const float* __restrict__ bias,
                                              _Float16* __restrict__ z2) {
  int wid = (blockIdx.x * blockDim.x + threadIdx.x) >> 5;
  int lane = threadIdx.x & 31;
  if (wid >= WN * 4 * 64) return;
  int n = wid / (4 * 64), r = wid - n * 4 * 64, mg = r / 64, tt = r - mg * 64;
  const _Float16* B = z1 + (size_t)n * 1024 * 256;
  v8f acc[4];
#pragma unroll
  for (int j = 0; j < 4; ++j) acc[j] = (v8f){};
#pragma unroll
  for (int kc = 0; kc < 256; kc += 32) {
    v16h b = load_b_frag(B, 256, tt * 16, kc, lane, 1024);
#pragma unroll
    for (int j = 0; j < 4; ++j)
      acc[j] = wmma_f16(load_a_frag(W, 256, (mg * 4 + j) * 16, kc, lane), b, acc[j]);
  }
  int t = tt * 16 + (lane & 15);
#pragma unroll
  for (int j = 0; j < 4; ++j) {
    int ob = d_obase((mg * 4 + j) * 16, lane);
    v8h o;
#pragma unroll
    for (int i = 0; i < 8; ++i) {
      float v = acc[j][i] + bias[ob + i];
      o[i] = (_Float16)(v > 0.0f ? v : 0.0f);
    }
    *(v8h*)(z2 + ((size_t)n * 1024 + t) * 256 + ob) = o;
  }
}

// ---- end2: tmp[t][oc] = z2 @ W2^T  (M=240 in 5-tile groups, K=256) --------

__global__ void __launch_bounds__(256) k_end2(const _Float16* __restrict__ z2,
                                              const _Float16* __restrict__ W,
                                              float* __restrict__ tmp) {
  int wid = (blockIdx.x * blockDim.x + threadIdx.x) >> 5;
  int lane = threadIdx.x & 31;
  if (wid >= WN * 3 * 64) return;
  int n = wid / (3 * 64), r = wid - n * 3 * 64, mg = r / 64, tt = r - mg * 64;
  const _Float16* B = z2 + (size_t)n * 1024 * 256;
  v8f acc[5];
#pragma unroll
  for (int j = 0; j < 5; ++j) acc[j] = (v8f){};
#pragma unroll
  for (int kc = 0; kc < 256; kc += 32) {
    v16h b = load_b_frag(B, 256, tt * 16, kc, lane, 1024);
#pragma unroll
    for (int j = 0; j < 5; ++j)
      acc[j] = wmma_f16(load_a_frag(W, 256, (mg * 5 + j) * 16, kc, lane), b, acc[j]);
  }
  int t = tt * 16 + (lane & 15);
#pragma unroll
  for (int j = 0; j < 5; ++j) {
    float* dst = tmp + ((size_t)n * 1024 + t) * 240 + d_obase((mg * 5 + j) * 16, lane);
#pragma unroll
    for (int i = 0; i < 8; ++i) dst[i] = acc[j][i];
  }
}

// ---- host orchestration ---------------------------------------------------

extern "C" void kernel_launch(void* const* d_in, const int* in_sizes, int n_in,
                              void* d_out, int out_size, void* d_ws, size_t ws_size,
                              hipStream_t stream) {
  const float* x   = (const float*)d_in[0];
  const float* sw  = (const float*)d_in[1];
  const float* fw  = (const float*)d_in[2];
  const float* gw  = (const float*)d_in[3];
  const float* rw  = (const float*)d_in[4];
  const float* skw = (const float*)d_in[5];
  const float* e1w = (const float*)d_in[6];
  const float* e1b = (const float*)d_in[7];
  const float* e2w = (const float*)d_in[8];
  const float* e2b = (const float*)d_in[9];

  char* base = (char*)d_ws;
  size_t off = 0;
  auto alloc = [&](size_t bytes) -> void* {
    void* p = base + off;
    off = (off + bytes + 255) & ~(size_t)255;
    return p;
  };
  _Float16* xT   = (_Float16*)alloc((size_t)WN * TCAP * 256 * 2);
  _Float16* hA   = (_Float16*)alloc((size_t)WN * TCAP * 64 * 2);
  _Float16* hB   = (_Float16*)alloc((size_t)WN * TCAP * 64 * 2);
  float*    skip = (float*)   alloc((size_t)WN * 1024 * 256 * 4);
  _Float16* z1   = (_Float16*)alloc((size_t)WN * 1024 * 256 * 2);
  _Float16* z2   = (_Float16*)alloc((size_t)WN * 1024 * 256 * 2);
  float*    tmp2 = (float*)   alloc((size_t)WN * 1024 * 240 * 4);
  _Float16* wS   = (_Float16*)alloc((size_t)64 * 256 * 2);
  _Float16* wF   = (_Float16*)alloc((size_t)40 * 2 * 64 * 64 * 2);
  _Float16* wG   = (_Float16*)alloc((size_t)40 * 2 * 64 * 64 * 2);
  _Float16* wR   = (_Float16*)alloc((size_t)40 * 64 * 64 * 2);
  _Float16* wK   = (_Float16*)alloc((size_t)40 * 256 * 64 * 2);
  _Float16* wE1  = (_Float16*)alloc((size_t)256 * 256 * 2);
  _Float16* wE2  = (_Float16*)alloc((size_t)240 * 256 * 2);

  const int TPB = 256;                       // 8 wave32 waves per block
  auto eg = [](int n) { return (n + 255) / 256; };
  auto wg = [](int waves) { return (waves * 32 + 255) / 256; };

  // per-launch weight conversion + input transpose + skip zeroing
  k_convert<<<eg(64 * 256), TPB, 0, stream>>>(sw, wS, 64 * 256);
  k_convert_tap<<<eg(40 * 2 * 64 * 64), TPB, 0, stream>>>(fw, wF, 40 * 2 * 64 * 64);
  k_convert_tap<<<eg(40 * 2 * 64 * 64), TPB, 0, stream>>>(gw, wG, 40 * 2 * 64 * 64);
  k_convert<<<eg(40 * 64 * 64), TPB, 0, stream>>>(rw, wR, 40 * 64 * 64);
  k_convert<<<eg(40 * 256 * 64), TPB, 0, stream>>>(skw, wK, 40 * 256 * 64);
  k_convert<<<eg(256 * 256), TPB, 0, stream>>>(e1w, wE1, 256 * 256);
  k_convert<<<eg(240 * 256), TPB, 0, stream>>>(e2w, wE2, 240 * 256);
  k_xpose<<<eg(WN * 256 * TCAP), TPB, 0, stream>>>(x, xT);
  k_zero<<<eg(WN * 1024 * 256), TPB, 0, stream>>>(skip, WN * 1024 * 256);

  // start 1x1 conv over all 8192 timesteps
  k_start<<<wg(WN * (TCAP / 16)), TPB, 0, stream>>>(xT, wS, hA);

  // 40 fused dilated gated layers (direct original-time form of dilate())
  int T = TCAP;
  _Float16* hcur = hA;
  _Float16* hnext = hB;
  int li = 0;
  for (int b = 0; b < 4; ++b) {
    int d = 1;
    for (int l = 0; l < 10; ++l) {
      int Tout = T - d;
      int TT = (Tout + 15) / 16;
      k_layer<<<wg(WN * TT), TPB, 0, stream>>>(
          hcur,
          wF + (size_t)li * 2 * 64 * 64, wG + (size_t)li * 2 * 64 * 64,
          wR + (size_t)li * 64 * 64, wK + (size_t)li * 256 * 64,
          hnext, skip, d, Tout, TT, Tout - 1024);
      _Float16* tsw = hcur; hcur = hnext; hnext = tsw;
      T = Tout;
      d *= 2;
      ++li;
    }
  }

  // end stages on the last 1024 timesteps only
  k_relu_h<<<eg(WN * 1024 * 256), TPB, 0, stream>>>(skip, z1, WN * 1024 * 256);
  k_end1<<<wg(WN * 4 * 64), TPB, 0, stream>>>(z1, wE1, e1b, z2);
  k_end2<<<wg(WN * 3 * 64), TPB, 0, stream>>>(z2, wE2, tmp2);
  k_final<<<eg(WN * 240 * 1024), TPB, 0, stream>>>(tmp2, e2b, (float*)d_out);

  (void)in_sizes; (void)n_in; (void)out_size; (void)ws_size;
}
